// SelfAttention_36558761623964
// MI455X (gfx1250) — matrix operand order repaired and checked by reading, not compiled, not run on hardware
//
#include <hip/hip_runtime.h>

typedef __bf16 bf16;
typedef __attribute__((ext_vector_type(16))) __bf16 v16bf;
typedef __attribute__((ext_vector_type(8)))  __bf16 v8bf;
typedef __attribute__((ext_vector_type(4)))  __bf16 v4bf;
typedef __attribute__((ext_vector_type(8)))  float  v8f;

#define D_EMBED 1024
#define N_HEADS 16
#define D_HEAD  64
#define SEQ     2048
#define BATCH   4

__device__ __forceinline__ v8f wmma_bf16(v16bf a, v16bf b, v8f c) {
  return __builtin_amdgcn_wmma_f32_16x16x32_bf16(
      false, a, false, b, (short)0, c, false, false);
}
__device__ __forceinline__ v16bf cat8(v8bf lo, v8bf hi) {
  return __builtin_shufflevector(lo, hi, 0, 1, 2, 3, 4, 5, 6, 7,
                                 8, 9, 10, 11, 12, 13, 14, 15);
}

// ---------------------------------------------------------------------------
// One-shot fp32 -> bf16 conversion (n divisible by 1024).
// ---------------------------------------------------------------------------
__global__ __launch_bounds__(256) void convert_f32_bf16(
    const float* __restrict__ s, bf16* __restrict__ d) {
  const size_t i = ((size_t)blockIdx.x * 256 + threadIdx.x) * 4;
  const float4 f = *(const float4*)(s + i);
  v4bf o;
  o[0] = (bf16)f.x; o[1] = (bf16)f.y; o[2] = (bf16)f.z; o[3] = (bf16)f.w;
  *(v4bf*)(d + i) = o;
}

// ---------------------------------------------------------------------------
// GEMM: C[M,N] = A[M,K]*B[K,N] + bias[N], A/B bf16, fp32 accum.
// Block tile 128x128, 8 waves (4xM, 2xN), wave tile 32x64 -> 8 WMMA / K-step.
// Double-buffered LDS: A tile staged with global_load_async_to_lds_b128
// (ASYNCcnt engine copy), B tile pulled to registers before the WMMAs and
// scattered (transposed) after them, so staging of tile k+1 overlaps the
// matrix work of tile k.
// ---------------------------------------------------------------------------
template <typename OT>
__global__ __launch_bounds__(256) void gemm_bias_bf16wmma(
    const bf16* __restrict__ A, const bf16* __restrict__ Bw,
    const float* __restrict__ bias, OT* __restrict__ C,
    int M, int N, int K) {
  __shared__ bf16 sA[2][128][40];    // [buf][m][k] row stride 80B (16B align)
  __shared__ bf16 sBt[2][128][40];   // [buf][n][k]

  const int tid  = threadIdx.x;
  const int lane = tid & 31;
  const int wid  = tid >> 5;
  const int wy   = wid & 3;            // 0..3 (M)
  const int wx   = wid >> 2;           // 0..1 (N)
  const int kh   = lane >> 4;
  const int l15  = lane & 15;

  const int block_m = blockIdx.x * 128;
  const int block_n = blockIdx.y * 128;

  v8f acc[2][4] = {};

  // ---- prologue: stage K-tile 0 into buffer 0 ----
#pragma unroll
  for (int it = 0; it < 2; ++it) {
    const int chunk = tid * 2 + it;              // 0..511
    const int r = chunk >> 2, q = (chunk & 3) * 8;
    const bf16* g = A + (size_t)(block_m + r) * K + q;
    const unsigned lds = (unsigned)(unsigned long long)&sA[0][r][q];
    asm volatile("global_load_async_to_lds_b128 %0, %1, off"
                 :: "v"(lds), "v"(g) : "memory");
  }
#pragma unroll
  for (int it = 0; it < 2; ++it) {
    const int chunk = tid + it * 256;            // 0..511
    const int r = chunk >> 4, c = (chunk & 15) * 8;
    const v8bf d = *(const v8bf*)(Bw + (size_t)r * N + block_n + c);
#pragma unroll
    for (int j = 0; j < 8; ++j) sBt[0][c + j][r] = d[j];
  }
  asm volatile("s_wait_asynccnt 0x0" ::: "memory");
  __syncthreads();

  const int KT = K >> 5;
  for (int kt = 0; kt < KT; ++kt) {
    const int cur = kt & 1, nxt = cur ^ 1;
    const bool has_next = (kt + 1) < KT;

    // ---- kick off staging of tile kt+1 (async A, B -> registers) ----
    v8bf bstage[2];
    if (has_next) {
      const int k0n = (kt + 1) * 32;
#pragma unroll
      for (int it = 0; it < 2; ++it) {
        const int chunk = tid * 2 + it;
        const int r = chunk >> 2, q = (chunk & 3) * 8;
        const bf16* g = A + (size_t)(block_m + r) * K + k0n + q;
        const unsigned lds = (unsigned)(unsigned long long)&sA[nxt][r][q];
        asm volatile("global_load_async_to_lds_b128 %0, %1, off"
                     :: "v"(lds), "v"(g) : "memory");
      }
#pragma unroll
      for (int it = 0; it < 2; ++it) {
        const int chunk = tid + it * 256;
        const int r = chunk >> 4, c = (chunk & 15) * 8;
        bstage[it] = *(const v8bf*)(Bw + (size_t)(k0n + r) * N + block_n + c);
      }
    }

    // ---- compute on buffer `cur` ----
    v16bf afr[2], bfr[4];
#pragma unroll
    for (int i = 0; i < 2; ++i) {
      const bf16* p = &sA[cur][wy * 32 + i * 16 + l15][kh * 8];
      afr[i] = cat8(*(const v8bf*)p, *(const v8bf*)(p + 16));
    }
#pragma unroll
    for (int j = 0; j < 4; ++j) {
      const bf16* p = &sBt[cur][wx * 64 + j * 16 + l15][kh * 8];
      bfr[j] = cat8(*(const v8bf*)p, *(const v8bf*)(p + 16));
    }
#pragma unroll
    for (int i = 0; i < 2; ++i)
#pragma unroll
      for (int j = 0; j < 4; ++j)
        acc[i][j] = wmma_bf16(afr[i], bfr[j], acc[i][j]);

    // ---- finish staging tile kt+1: scatter B transposed ----
    if (has_next) {
#pragma unroll
      for (int it = 0; it < 2; ++it) {
        const int chunk = tid + it * 256;
        const int r = chunk >> 4, c = (chunk & 15) * 8;
#pragma unroll
        for (int j = 0; j < 8; ++j) sBt[nxt][c + j][r] = bstage[it][j];
      }
    }
    asm volatile("s_wait_asynccnt 0x0" ::: "memory");
    __syncthreads();
  }

  // ---- epilogue: +bias, convert, store ----
#pragma unroll
  for (int i = 0; i < 2; ++i)
#pragma unroll
    for (int j = 0; j < 4; ++j) {
      const int ncol = block_n + wx * 64 + j * 16 + l15;
      const float bv = bias[ncol];
#pragma unroll
      for (int r = 0; r < 8; ++r) {
        const int mrow = block_m + wy * 32 + i * 16 + r + 8 * kh;
        C[(size_t)mrow * N + ncol] = (OT)(acc[i][j][r] + bv);
      }
    }
}

// ---------------------------------------------------------------------------
// Causal flash attention. qkv bf16 [B*S, 3D] -> out bf16 [B*S, D].
// One wave owns 16 q rows. Softmax denominator is maintained by an extra
// ones-column WMMA (same corr rescaling as the output accumulator), so the
// only cross-lane reduction left per key block is the running-max.
// ---------------------------------------------------------------------------
__global__ __launch_bounds__(256) void attn_flash_bf16wmma(
    const bf16* __restrict__ qkv, bf16* __restrict__ out) {
  const int b  = blockIdx.z;
  const int h  = blockIdx.y;
  const int qt = blockIdx.x;
  const int tid  = threadIdx.x;
  const int lane = tid & 31;
  const int wid  = tid >> 5;
  const int kh   = lane >> 4;
  const int l15  = lane & 15;

  const int D3 = 3 * D_EMBED;
  const size_t rowbase = (size_t)b * SEQ * D3;
  const int qcol = h * D_HEAD;
  const int kcol = D_EMBED + h * D_HEAD;
  const int vcol = 2 * D_EMBED + h * D_HEAD;

  const int q0 = qt * 128 + wid * 16;

  __shared__ bf16 sP[8][16][40];    // per-wave P transpose
  __shared__ bf16 sVt[8][64][40];   // per-wave V^T [dh][key_local]

  // Q as A-fragments, pre-scaled by 1/sqrt(64)=2^-3 (exact in bf16)
  v16bf qfr[2];
  {
    const bf16* qrow = qkv + rowbase + (size_t)(q0 + l15) * D3 + qcol;
#pragma unroll
    for (int t = 0; t < 2; ++t) {
      const bf16* p = qrow + t * 32 + kh * 8;
      qfr[t] = cat8(*(const v8bf*)p, *(const v8bf*)(p + 16));
#pragma unroll
      for (int e = 0; e < 16; ++e)
        qfr[t][e] = (bf16)((float)qfr[t][e] * 0.125f);
    }
  }

  // constant ones-column B fragment: column n==0 is 1, others 0
  v16bf onesfr;
  {
    const bf16 ov = (bf16)((l15 == 0) ? 1.0f : 0.0f);
#pragma unroll
    for (int e = 0; e < 16; ++e) onesfr[e] = ov;
  }

  v8f acc[4] = {};
  v8f acc_l  = {};                 // softmax denominator (column 0)
  float rm[8];
#pragma unroll
  for (int r = 0; r < 8; ++r) rm[r] = -1e30f;

  const int qmax = q0 + 15;

  for (int kb = 0; kb <= qmax; kb += 32) {
    // ---- stage V^T tile (32 keys x 64 dh) for this wave ----
    const bf16* vbase = qkv + rowbase + vcol;
#pragma unroll
    for (int i = 0; i < 8; ++i) {
      const int chunk = i * 32 + lane;
      const int row = chunk >> 3;            // key-local 0..31
      const int cq  = (chunk & 7) * 8;       // dh chunk
      const v8bf d = *(const v8bf*)(vbase + (size_t)(kb + row) * D3 + cq);
#pragma unroll
      for (int j = 0; j < 8; ++j) sVt[wid][cq + j][row] = d[j];
    }

    // ---- scores S = (Q/8)*K^T (two 16x16 C tiles over 32 keys) ----
    v8f sfr[2];
#pragma unroll
    for (int j = 0; j < 2; ++j) {
      const int key = kb + j * 16 + l15;
      const bf16* krow = qkv + rowbase + (size_t)key * D3 + kcol;
      v8f s = {};
#pragma unroll
      for (int t = 0; t < 2; ++t) {
        const bf16* p = krow + t * 32 + kh * 8;
        const v16bf kfr = cat8(*(const v8bf*)p, *(const v8bf*)(p + 16));
        s = wmma_bf16(qfr[t], kfr, s);
      }
      sfr[j] = s;
    }

    // ---- causal mask (wave-uniform skip for interior tiles) ----
    if (kb + 31 > q0) {
#pragma unroll
      for (int j = 0; j < 2; ++j) {
        const int key = kb + j * 16 + l15;
#pragma unroll
        for (int r = 0; r < 8; ++r) {
          const int q = q0 + r + 8 * kh;
          sfr[j][r] = (key > q) ? -1e30f : sfr[j][r];
        }
      }
    }

    // ---- running max (batched xor-shuffle over the 16-lane row group) ----
    float tmax[8];
#pragma unroll
    for (int r = 0; r < 8; ++r) tmax[r] = fmaxf(sfr[0][r], sfr[1][r]);
#pragma unroll
    for (int off = 1; off < 16; off <<= 1) {
      float t[8];
#pragma unroll
      for (int r = 0; r < 8; ++r) t[r] = __shfl_xor(tmax[r], off, 32);
#pragma unroll
      for (int r = 0; r < 8; ++r) tmax[r] = fmaxf(tmax[r], t[r]);
    }

    float corr[8];
#pragma unroll
    for (int r = 0; r < 8; ++r) {
      const float mnew = fmaxf(rm[r], tmax[r]);
      corr[r] = __expf(rm[r] - mnew);
      rm[r] = mnew;
    }
#pragma unroll
    for (int j = 0; j < 2; ++j)
#pragma unroll
      for (int r = 0; r < 8; ++r)
        sfr[j][r] = __expf(sfr[j][r] - rm[r]);

    // rescale output accumulator AND denominator accumulator
#pragma unroll
    for (int t = 0; t < 4; ++t)
#pragma unroll
      for (int r = 0; r < 8; ++r) acc[t][r] *= corr[r];
#pragma unroll
    for (int r = 0; r < 8; ++r) acc_l[r] *= corr[r];

    // ---- P: C-layout -> LDS -> A-fragment (per-wave, DS in-order) ----
#pragma unroll
    for (int j = 0; j < 2; ++j)
#pragma unroll
      for (int r = 0; r < 8; ++r)
        sP[wid][r + 8 * kh][j * 16 + l15] = (bf16)sfr[j][r];

    v16bf pfr;
    {
      const bf16* p = &sP[wid][l15][kh * 8];
      pfr = cat8(*(const v8bf*)p, *(const v8bf*)(p + 16));
    }

    // ---- acc += P*V ; acc_l += P*ones (denominator) ----
#pragma unroll
    for (int t = 0; t < 4; ++t) {
      const bf16* p = &sVt[wid][t * 16 + l15][kh * 8];
      const v16bf vfr = cat8(*(const v8bf*)p, *(const v8bf*)(p + 16));
      acc[t] = wmma_bf16(pfr, vfr, acc[t]);
    }
    acc_l = wmma_bf16(pfr, onesfr, acc_l);
  }

  // ---- broadcast denominators (column 0 lives in lane kh*16), normalize ----
  float rinv[8];
#pragma unroll
  for (int r = 0; r < 8; ++r) {
    const float s = __shfl(acc_l[r], kh * 16, 32);
    rinv[r] = 1.0f / s;
  }
#pragma unroll
  for (int t = 0; t < 4; ++t)
#pragma unroll
    for (int r = 0; r < 8; ++r) {
      const int q = q0 + r + 8 * kh;
      out[((size_t)b * SEQ + q) * D_EMBED + h * D_HEAD + t * 16 + l15] =
          (bf16)(acc[t][r] * rinv[r]);
    }
}

// ---------------------------------------------------------------------------
extern "C" void kernel_launch(void* const* d_in, const int* in_sizes, int n_in,
                              void* d_out, int out_size, void* d_ws,
                              size_t ws_size, hipStream_t stream) {
  const float* x     = (const float*)d_in[0];  // [4,2048,1024]
  const float* w_in  = (const float*)d_in[1];  // [1024,3072]
  const float* b_in  = (const float*)d_in[2];  // [3072]
  const float* w_out = (const float*)d_in[3];  // [1024,1024]
  const float* b_out = (const float*)d_in[4];  // [1024]

  const int M  = BATCH * SEQ;                  // 8192
  const size_t n_x  = (size_t)M * D_EMBED;
  const size_t n_wi = (size_t)D_EMBED * 3 * D_EMBED;
  const size_t n_wo = (size_t)D_EMBED * D_EMBED;

  char* ws = (char*)d_ws;
  bf16* xb   = (bf16*)ws;                 ws += n_x * 2;
  bf16* wib  = (bf16*)ws;                 ws += n_wi * 2;
  bf16* wob  = (bf16*)ws;                 ws += n_wo * 2;
  bf16* qkv  = (bf16*)ws;                 ws += (size_t)M * 3 * D_EMBED * 2;
  bf16* attn = (bf16*)ws;

  dim3 blk(256);

  // 0) one-shot fp32 -> bf16 conversions
  convert_f32_bf16<<<dim3(n_x  / 1024), blk, 0, stream>>>(x, xb);
  convert_f32_bf16<<<dim3(n_wi / 1024), blk, 0, stream>>>(w_in, wib);
  convert_f32_bf16<<<dim3(n_wo / 1024), blk, 0, stream>>>(w_out, wob);

  // 1) QKV projection: [8192,1024] x [1024,3072] + b_in -> bf16
  gemm_bias_bf16wmma<bf16>
      <<<dim3(M / 128, (3 * D_EMBED) / 128), blk, 0, stream>>>(
          xb, wib, b_in, qkv, M, 3 * D_EMBED, D_EMBED);

  // 2) causal flash attention -> bf16 [8192,1024]
  attn_flash_bf16wmma<<<dim3(SEQ / 128, N_HEADS, BATCH), blk, 0, stream>>>(
      qkv, attn);

  // 3) output projection: [8192,1024] x [1024,1024] + b_out -> fp32 d_out
  gemm_bias_bf16wmma<float>
      <<<dim3(M / 128, D_EMBED / 128), blk, 0, stream>>>(
          attn, wob, b_out, (float*)d_out, M, D_EMBED, D_EMBED);

  (void)in_sizes; (void)n_in; (void)out_size; (void)ws_size;
}
